// spatialAttentionGCN_74603581931766
// MI455X (gfx1250) — matrix operand order repaired and checked by reading, not compiled, not run on hardware
//
#include <hip/hip_runtime.h>
#include <math.h>

#define B_ 8
#define N_ 2048
#define C_ 64
#define NT16 (N_ / 16)   // 128 tiles of 16
#define JH 1024          // j-panel depth per phase in the fused G kernel (32KB LDS)

typedef __attribute__((ext_vector_type(16))) __bf16 v16bf;
typedef __attribute__((ext_vector_type(8)))  float  v8f;
typedef __attribute__((ext_vector_type(4)))  unsigned v4u;
typedef __attribute__((ext_vector_type(8)))  int   v8i;
typedef __attribute__((ext_vector_type(4)))  int   v4i;

#if defined(__has_builtin)
#if __has_builtin(__builtin_amdgcn_tensor_load_to_lds) && \
    __has_builtin(__builtin_amdgcn_s_wait_tensorcnt)
#define HAVE_TDM 1
#endif
#endif

union FragU { uint4 u[2]; v16bf v; };
union VecU  { uint4 u;    __bf16 h[8]; };

// ---------------------------------------------------------------------------
// WMMA fragment loaders (CDNA5 16-bit layouts, wave32).
// A-matrix 16x32 (MxK): lane m=lane&15, half kh=lane>>4.
//   elements 0..7  = K in [kh*8,    kh*8+8)      (16B contiguous)
//   elements 8..15 = K in [16+kh*8, 16+kh*8+8)   (16B contiguous)
// B-matrix 32x16 (KxN) from a transposed array Bt[n][k]:
//   lane n=lane&15, same K order as A: [kh*8,+8) then [16+kh*8,+8)
// ---------------------------------------------------------------------------
__device__ __forceinline__ v16bf load_a_frag(const __bf16* base, int lda,
                                             int row0, int k0, int lane) {
  int m = lane & 15, kh = lane >> 4;
  const __bf16* p = base + (size_t)(row0 + m) * lda + (k0 + kh * 8);
  FragU f;
  f.u[0] = *(const uint4*)p;
  f.u[1] = *(const uint4*)(p + 16);
  return f.v;
}

__device__ __forceinline__ v16bf load_b_frag(const __bf16* bt, int ldb,
                                             int n0, int k0, int lane) {
  int n = lane & 15, klo = (lane >> 4) * 8;
  const __bf16* p = bt + (size_t)(n0 + n) * ldb + (k0 + klo);
  FragU f;
  f.u[0] = *(const uint4*)p;
  f.u[1] = *(const uint4*)(p + 16);
  return f.v;
}

__device__ __forceinline__ v8f wmma_bf(v16bf a, v16bf b, v8f c) {
  // D = A*B + C, fp32 accumulate -> v_wmma_f32_16x16x32_bf16
  return __builtin_amdgcn_wmma_f32_16x16x32_bf16(false, a, false, b,
                                                 (short)0, c, false, false);
}

// ---------------------------------------------------------------------------
// Small prep kernels
// ---------------------------------------------------------------------------
__global__ void k_f32_to_bf16(const float* __restrict__ s, __bf16* __restrict__ d, int n) {
  int i = blockIdx.x * 256 + threadIdx.x;
  if (i < n) d[i] = (__bf16)s[i];
}

__global__ void k_transpose_bf16(const float* __restrict__ s, __bf16* __restrict__ d,
                                 int R, int Cc) {
  // s is RxCc row-major; d is CcxR row-major: d[c][r] = s[r][c]
  int i = blockIdx.x * 256 + threadIdx.x;
  if (i < R * Cc) {
    int r = i / Cc, c = i % Cc;
    d[(size_t)c * R + r] = (__bf16)s[i];
  }
}

__global__ void k_diag(const float* __restrict__ A, float* __restrict__ dg) {
  int i = blockIdx.x * 256 + threadIdx.x;
  if (i < N_) dg[i] = A[(size_t)i * N_ + i];
}

__global__ void k_zero_f32(float* __restrict__ p, int n) {
  int i = blockIdx.x * 256 + threadIdx.x;
  if (i < n) p[i] = 0.0f;
}

// ---------------------------------------------------------------------------
// Q = x@Wq, K = x@Wk (bf16 out). One block per (b, 16-row strip).
// Waves 0-3 -> Q col-tiles 0-3, waves 4-7 -> K col-tiles 0-3.
// Two independent accumulators over the K=64 depth for WMMA ILP.
// ---------------------------------------------------------------------------
__global__ __launch_bounds__(256) void k_qk_gemm(
    const __bf16* __restrict__ xbf, const __bf16* __restrict__ WqT,
    const __bf16* __restrict__ WkT, __bf16* __restrict__ Qbf,
    __bf16* __restrict__ Kbf) {
  int bid = blockIdx.x;
  int b = bid / NT16, mt = bid % NT16, row0 = mt * 16;
  int tid = threadIdx.x, wave = tid >> 5, lane = tid & 31;
  const __bf16* xb = xbf + (size_t)b * N_ * C_;
  const __bf16* Bt = (wave < 4) ? WqT : WkT;
  __bf16* dst = ((wave < 4) ? Qbf : Kbf) + (size_t)b * N_ * C_;
  int col0 = (wave & 3) * 16;
  v8f acc0 = {}, acc1 = {};
  acc0 = wmma_bf(load_a_frag(xb, C_, row0, 0, lane),
                 load_b_frag(Bt, C_, col0, 0, lane), acc0);
  acc1 = wmma_bf(load_a_frag(xb, C_, row0, 32, lane),
                 load_b_frag(Bt, C_, col0, 32, lane), acc1);
  int col = col0 + (lane & 15), r0 = (lane >> 4) * 8;
  for (int v = 0; v < 8; ++v)
    dst[(size_t)(row0 + r0 + v) * C_ + col] = (__bf16)(acc0[v] + acc1[v]);
}

// ---------------------------------------------------------------------------
// V[b] = x[b] @ Wv  -> (N x N) bf16.  B-operand is pre-transposed WvT[n][c].
// ---------------------------------------------------------------------------
__global__ __launch_bounds__(256) void k_v_gemm(
    const __bf16* __restrict__ xbf, const __bf16* __restrict__ WvT,
    __bf16* __restrict__ Vbf) {
  int bid = blockIdx.x;
  int b = bid / NT16, mt = bid % NT16, row0 = mt * 16;
  int tid = threadIdx.x, wave = tid >> 5, lane = tid & 31;
  const __bf16* xb = xbf + (size_t)b * N_ * C_;
  __bf16* vout = Vbf + (size_t)b * N_ * N_;
  int col = lane & 15, r0 = (lane >> 4) * 8;
  // A-fragments invariant across n-tiles: hoist.
  v16bf a0 = load_a_frag(xb, C_, row0, 0, lane);
  v16bf a1 = load_a_frag(xb, C_, row0, 32, lane);
  for (int nt = wave; nt < NT16; nt += 8) {
    int col0 = nt * 16;
    v8f acc0 = {}, acc1 = {};
    acc0 = wmma_bf(a0, load_b_frag(WvT, C_, col0, 0, lane), acc0);
    acc1 = wmma_bf(a1, load_b_frag(WvT, C_, col0, 32, lane), acc1);
    for (int v = 0; v < 8; ++v)
      vout[(size_t)(row0 + r0 + v) * N_ + col0 + col] = (__bf16)(acc0[v] + acc1[v]);
  }
}

// ---------------------------------------------------------------------------
// St[b] = K[b] @ Q[b]^T  (this is score^T: St[m][j] = Q[j]·K[m]).
// fp16 raw logits out (softmax axis=1 of score == row softmax of St).
// ---------------------------------------------------------------------------
__global__ __launch_bounds__(256) void k_st_gemm(
    const __bf16* __restrict__ Kbf, const __bf16* __restrict__ Qbf,
    _Float16* __restrict__ St) {
  int bid = blockIdx.x;
  int b = bid / NT16, mt = bid % NT16, row0 = mt * 16;
  int tid = threadIdx.x, wave = tid >> 5, lane = tid & 31;
  const __bf16* kb = Kbf + (size_t)b * N_ * C_;
  const __bf16* qb = Qbf + (size_t)b * N_ * C_;
  _Float16* sout = St + (size_t)b * N_ * N_;
  int col = lane & 15, r0 = (lane >> 4) * 8;
  v16bf a0 = load_a_frag(kb, C_, row0, 0, lane);
  v16bf a1 = load_a_frag(kb, C_, row0, 32, lane);
  for (int jt = wave; jt < NT16; jt += 8) {
    int j0 = jt * 16;
    v8f acc0 = {}, acc1 = {};
    acc0 = wmma_bf(a0, load_b_frag(qb, C_, j0, 0, lane), acc0);
    acc1 = wmma_bf(a1, load_b_frag(qb, C_, j0, 32, lane), acc1);
    for (int v = 0; v < 8; ++v)
      sout[(size_t)(row0 + r0 + v) * N_ + j0 + col] = (_Float16)(acc0[v] + acc1[v]);
  }
}

// ---------------------------------------------------------------------------
// Row softmax of St -> bf16 score^T.  One block per (b, m) row of 2048.
// ---------------------------------------------------------------------------
__global__ __launch_bounds__(256) void k_softmax_rows(
    const _Float16* __restrict__ St, __bf16* __restrict__ Stbf) {
  int row = blockIdx.x, t = threadIdx.x;
  const _Float16* src = St + (size_t)row * N_;
  __bf16* dst = Stbf + (size_t)row * N_;
  __shared__ float red[256];
  float vals[8];
  float mx = -3.0e38f;
  for (int i = 0; i < 8; ++i) {
    vals[i] = (float)src[t + i * 256];
    mx = fmaxf(mx, vals[i]);
  }
  red[t] = mx; __syncthreads();
  for (int s = 128; s > 0; s >>= 1) {
    if (t < s) red[t] = fmaxf(red[t], red[t + s]);
    __syncthreads();
  }
  mx = red[0]; __syncthreads();
  float sum = 0.0f;
  for (int i = 0; i < 8; ++i) { vals[i] = __expf(vals[i] - mx); sum += vals[i]; }
  red[t] = sum; __syncthreads();
  for (int s = 128; s > 0; s >>= 1) {
    if (t < s) red[t] += red[t + s];
    __syncthreads();
  }
  float inv = 1.0f / red[0];
  for (int i = 0; i < 8; ++i) dst[t + i * 256] = (__bf16)(vals[i] * inv);
}

// ---------------------------------------------------------------------------
// Fused dominant GEMM:  dy[b,n] = sum_m (A @ score[b])[n,m] * V[b,m,n]
//                               = sum_m sum_j A[n,j] * St[b,m,j] * V[b,m,n]
// One block per (b, 16-n strip). A-panel (16 x JH bf16, 32KB) staged in LDS
// via the Tensor Data Mover when available (D# descriptor, s_wait_tensorcnt);
// two j-phases, each contributing a partial-G tile -> partial dy (linear in G).
// Inner K-loop uses two independent WMMA accumulators for ILP.
// ---------------------------------------------------------------------------
__global__ __launch_bounds__(256) void k_g_dy(
    const __bf16* __restrict__ Abf, const __bf16* __restrict__ Stbf,
    const __bf16* __restrict__ Vbf, float* __restrict__ dy) {
  int bid = blockIdx.x;
  int b = bid / NT16, nt = bid % NT16, n0 = nt * 16;
  int tid = threadIdx.x, wave = tid >> 5, lane = tid & 31;
  __shared__ __bf16 Apan[16 * JH];  // 32KB
  const __bf16* stb = Stbf + (size_t)b * N_ * N_;
  const __bf16* vb  = Vbf  + (size_t)b * N_ * N_;
  int col = lane & 15, r0 = (lane >> 4) * 8;
  float dyp[8];
  for (int v = 0; v < 8; ++v) dyp[v] = 0.0f;

  for (int half = 0; half < 2; ++half) {
    __syncthreads();  // previous phase's readers done before overwrite
#ifdef HAVE_TDM
    if (wave == 0) {
      // TDM descriptor: 2D tile, 16 rows x 2048B (data_size=8B -> tile_dim0=256),
      // global row stride N_*2B = 4096B (512 units), into contiguous LDS.
      unsigned lds_base = (unsigned)(size_t)(void*)Apan;
      unsigned long long ga =
          (unsigned long long)(size_t)(Abf + (size_t)n0 * N_ + half * JH);
      v4u g0;
      g0[0] = 1u;                                   // count=1, user descriptor
      g0[1] = lds_base;                             // lds_addr
      g0[2] = (unsigned)(ga & 0xFFFFFFFFu);         // global_addr[31:0]
      g0[3] = (unsigned)((ga >> 32) & 0x01FFFFFFu)  // global_addr[56:32]
              | (2u << 30);                         // type=2 (image)
      v8i g1;
      g1[0] = (int)(3u << 16);     // workgroup_mask=0, data_size=3 (8B)
      g1[1] = (int)(512u << 16);   // tensor_dim0[15:0] = 512 units
      g1[2] = (int)(16u << 16);    // tensor_dim0 hi=0 | tensor_dim1[15:0]=16
      g1[3] = (int)(256u << 16);   // tensor_dim1 hi=0 | tile_dim0=256
      g1[4] = 16;                  // tile_dim1=16, tile_dim2=0
      g1[5] = 512;                 // tensor_dim0_stride[31:0] = 512 units
      g1[6] = 0;                   // stride hi / tensor_dim1_stride lo
      g1[7] = 0;
      v4i g2 = {0, 0, 0, 0}, g3 = {0, 0, 0, 0};
      v8i g4 = {0, 0, 0, 0, 0, 0, 0, 0};  // extra group (clang-23 6-arg form)
      __builtin_amdgcn_tensor_load_to_lds(g0, g1, g2, g3, g4, 0);
      __builtin_amdgcn_s_wait_tensorcnt(0);
    }
#else
    // cooperative fallback: A rows n0..n0+15, j in [half*JH, half*JH+JH)
    for (int i = tid; i < 16 * JH / 8; i += 256) {
      int r = i / (JH / 8), cc = i % (JH / 8);
      ((uint4*)Apan)[i] =
          *(const uint4*)(Abf + (size_t)(n0 + r) * N_ + half * JH + cc * 8);
    }
#endif
    __syncthreads();

    for (int mt = wave; mt < NT16; mt += 8) {
      int m0 = mt * 16;
      v8f acc0 = {}, acc1 = {};
      for (int jt = 0; jt < JH / 32; jt += 2) {
        if (jt + 2 < JH / 32)
          __builtin_prefetch(stb + (size_t)(m0 + col) * N_ + half * JH +
                                 (jt + 2) * 32, 0, 0);
        v16bf a0 = load_a_frag(Apan, JH, 0, jt * 32, lane);
        v16bf b0 = load_b_frag(stb, N_, m0, half * JH + jt * 32, lane);
        v16bf a1 = load_a_frag(Apan, JH, 0, (jt + 1) * 32, lane);
        v16bf b1 = load_b_frag(stb, N_, m0, half * JH + (jt + 1) * 32, lane);
        acc0 = wmma_bf(a0, b0, acc0);
        acc1 = wmma_bf(a1, b1, acc1);
      }
      // epilogue: lane holds partial G[n0+r0+v, m0+col]; dot with V[m0+col, n0+r0+v]
      VecU vv;
      vv.u = *(const uint4*)(vb + (size_t)(m0 + col) * N_ + n0 + r0);
      for (int v = 0; v < 8; ++v) dyp[v] += (acc0[v] + acc1[v]) * (float)vv.h[v];
    }
  }
  for (int v = 0; v < 8; ++v)
    atomicAdd(&dy[(size_t)b * N_ + n0 + r0 + v], dyp[v]);
}

// ---------------------------------------------------------------------------
// out = relu(alpha * diag(A)[n] * x  +  beta * (dy[b,n]*(x_row @ theta_w) + theta_b))
// ---------------------------------------------------------------------------
__global__ __launch_bounds__(256) void k_final(
    const float* __restrict__ x, const float* __restrict__ Adiag,
    const float* __restrict__ dy, const float* __restrict__ theta_w,
    const float* __restrict__ theta_b, const float* __restrict__ alpha,
    const float* __restrict__ beta, float* __restrict__ out) {
  int idx = blockIdx.x * 256 + threadIdx.x;  // B*N*C total
  int row = idx >> 6, c = idx & 63, n = row & (N_ - 1);
  const float* xr = x + (size_t)row * C_;
  float dot = 0.0f;
  for (int i = 0; i < C_; ++i) dot = fmaf(xr[i], theta_w[i * C_ + c], dot);
  float val = alpha[0] * Adiag[n] * x[idx] + beta[0] * (dy[row] * dot + theta_b[c]);
  out[idx] = fmaxf(val, 0.0f);
}

// ---------------------------------------------------------------------------
extern "C" void kernel_launch(void* const* d_in, const int* in_sizes, int n_in,
                              void* d_out, int out_size, void* d_ws, size_t ws_size,
                              hipStream_t stream) {
  const float* x       = (const float*)d_in[0];
  const float* A       = (const float*)d_in[1];
  const float* Wq      = (const float*)d_in[2];
  const float* Wk      = (const float*)d_in[3];
  const float* Wv      = (const float*)d_in[4];
  const float* theta_w = (const float*)d_in[5];
  const float* theta_b = (const float*)d_in[6];
  const float* alpha   = (const float*)d_in[7];
  const float* beta    = (const float*)d_in[8];
  float* out = (float*)d_out;

  char* w = (char*)d_ws;
  size_t off = 0;
  auto alloc = [&](size_t bytes) -> char* {
    char* p = w + off;
    off += (bytes + 255) & ~(size_t)255;
    return p;
  };
  __bf16*   xbf   = (__bf16*)  alloc((size_t)B_ * N_ * C_ * 2);
  __bf16*   Abf   = (__bf16*)  alloc((size_t)N_ * N_ * 2);
  float*    Adiag = (float*)   alloc((size_t)N_ * 4);
  __bf16*   WqT   = (__bf16*)  alloc((size_t)C_ * C_ * 2);
  __bf16*   WkT   = (__bf16*)  alloc((size_t)C_ * C_ * 2);
  __bf16*   WvT   = (__bf16*)  alloc((size_t)N_ * C_ * 2);
  __bf16*   Qbf   = (__bf16*)  alloc((size_t)B_ * N_ * C_ * 2);
  __bf16*   Kbf   = (__bf16*)  alloc((size_t)B_ * N_ * C_ * 2);
  __bf16*   Vbf   = (__bf16*)  alloc((size_t)B_ * N_ * N_ * 2);
  _Float16* Sth   = (_Float16*)alloc((size_t)B_ * N_ * N_ * 2);
  __bf16*   Stbf  = (__bf16*)  alloc((size_t)B_ * N_ * N_ * 2);
  float*    dyb   = (float*)   alloc((size_t)B_ * N_ * 4);

  const int nx = B_ * N_ * C_;   // 1,048,576
  const int nA = N_ * N_;        // 4,194,304

  k_f32_to_bf16<<<(nx + 255) / 256, 256, 0, stream>>>(x, xbf, nx);
  k_f32_to_bf16<<<(nA + 255) / 256, 256, 0, stream>>>(A, Abf, nA);
  k_diag<<<(N_ + 255) / 256, 256, 0, stream>>>(A, Adiag);
  k_transpose_bf16<<<(C_ * C_ + 255) / 256, 256, 0, stream>>>(Wq, WqT, C_, C_);
  k_transpose_bf16<<<(C_ * C_ + 255) / 256, 256, 0, stream>>>(Wk, WkT, C_, C_);
  k_transpose_bf16<<<(C_ * N_ + 255) / 256, 256, 0, stream>>>(Wv, WvT, C_, N_);
  k_zero_f32<<<(B_ * N_ + 255) / 256, 256, 0, stream>>>(dyb, B_ * N_);

  dim3 blk(256);
  dim3 grd(B_ * NT16);  // 1024 blocks
  k_qk_gemm<<<grd, blk, 0, stream>>>(xbf, WqT, WkT, Qbf, Kbf);
  k_v_gemm<<<grd, blk, 0, stream>>>(xbf, WvT, Vbf);
  k_st_gemm<<<grd, blk, 0, stream>>>(Kbf, Qbf, Sth);
  k_softmax_rows<<<B_ * N_, 256, 0, stream>>>(Sth, Stbf);
  k_g_dy<<<grd, blk, 0, stream>>>(Abf, Stbf, Vbf, dyb);
  k_final<<<(B_ * N_ * C_) / 256, 256, 0, stream>>>(x, Adiag, dyb, theta_w,
                                                    theta_b, alpha, beta, out);
}
